// KnowledgePooling_80633716015133
// MI455X (gfx1250) — compile-verified, matrix operand order checked by reading, not run on hardware
//
#include <hip/hip_runtime.h>
#include <hip/hip_bf16.h>
#include <float.h>

// ---------------------------------------------------------------------------
// KnowledgePooling on MI455X (gfx1250, wave32).
//
//   q   = k_feature @ Wq.T                         [G, D]   (WMMA f16, K=2048)
//   p   = q @ Wk                                   [G, D]   (WMMA f16)
//   s_n = (x_n . p_{seg(n)}) / sqrt(D)             [N]      (streaming f32 VALU)
//   softmax per segment; R_g = sum_n attn_n x_n    [G, D]   (fused, f32)
//   out = R @ Wv.T + bv                            [G, D]   (WMMA f16; sum attn = 1)
//   gi  = out @ W_ih.T + b_ih ; gh = q @ W_hh.T + b_hh      (WMMA f16)
//   GRU gates -> d_out                             [G, D]   (f32 VALU)
//
// B operands pre-packed into WMMA-fragment order (one contiguous 32B read per
// lane per tile); A fragments are two contiguous 16B runs per lane. Each wave
// computes a 16x32 output strip (2 WMMAs share one A fragment).
// ---------------------------------------------------------------------------

#define Nn 131072
#define Gg 4096
#define Dd 256
#define FPd 2048

typedef __attribute__((ext_vector_type(16))) _Float16 v16h;
typedef __attribute__((ext_vector_type(8)))  _Float16 v8h;
typedef __attribute__((ext_vector_type(8)))  float    v8f;

// ---------------------------- conversions ----------------------------------

__global__ void cvt_f32_to_f16(const float* __restrict__ in, _Float16* __restrict__ out, int n) {
    int i = blockIdx.x * blockDim.x + threadIdx.x;
    if (i < n) out[i] = (_Float16)in[i];
}

// Pack a B operand (logical [K,N], B[k][n]) into WMMA-fragment order:
//   Bp[((kt*N + c)*2 + h)*16 + e]  <-  B[kt*32 + 8h + (e<8 ? e : e+8)][c]
// transposed=1: source W is [N,K] row-major (B[k][n] = W[n*K+k])
// transposed=0: source W is [K,N] row-major (B[k][n] = W[k*N+n])
__global__ void pack_b_f16(const float* __restrict__ W, _Float16* __restrict__ Bp,
                           int K, int N, int transposed) {
    int o = blockIdx.x * blockDim.x + threadIdx.x;
    if (o >= K * N) return;
    const int e   = o & 15;
    const int h   = (o >> 4) & 1;
    const int t   = o >> 5;
    const int c   = t % N;
    const int kt  = t / N;
    const int k   = kt * 32 + 8 * h + (e < 8 ? e : e + 8);
    const float v = transposed ? W[(size_t)c * K + k] : W[(size_t)k * N + c];
    Bp[o] = (_Float16)v;
}

// ---------------------------- WMMA GEMM ------------------------------------
// A [M,K] f16 row-major; Bp packed as above; C [M,N] f32 (+ optional bias[n]).
// One wave computes a 16x32 strip (two 16x16 tiles sharing the A fragment);
// 4 waves per block -> 16x128 per block. ISA 7.12.2 fragment layouts.

__global__ void wmma_gemm_f16_f32(const _Float16* __restrict__ A,
                                  const _Float16* __restrict__ Bp,
                                  const float* __restrict__ bias,
                                  float* __restrict__ C,
                                  int M, int N, int K) {
    const int lane = threadIdx.x & 31;
    const int wave = threadIdx.x >> 5;                       // 4 waves per block
    const int col0 = (blockIdx.x * 4 + wave) * 32;           // 32-wide strip
    const int row0 = blockIdx.y * 16;
    if (col0 >= N || row0 >= M) return;

    const int row  = row0 + (lane & 15);
    const int half = lane >> 4;

    // strength-reduced running pointers
    const _Float16* aptr = A + (size_t)row * K + 8 * half;                       // +=32 per kt
    const _Float16* bptr = Bp + ((size_t)((col0 + (lane & 15)) * 2 + half) << 4); // +=N*32 per kt
    const size_t bstep = (size_t)N * 32;

    v8f acc0 = {}, acc1 = {};
    const int KT = K >> 5;
    for (int kt = 0; kt < KT; ++kt) {
        const v8h lo = *(const v8h*)(aptr);
        const v8h hi = *(const v8h*)(aptr + 16);
        const v16h a = __builtin_shufflevector(lo, hi, 0, 1, 2, 3, 4, 5, 6, 7,
                                                       8, 9, 10, 11, 12, 13, 14, 15);
        const v16h b0 = *(const v16h*)(bptr);        // tile col0   .. col0+15
        const v16h b1 = *(const v16h*)(bptr + 512);  // tile col0+16.. col0+31
        acc0 = __builtin_amdgcn_wmma_f32_16x16x32_f16(false, a, false, b0,
                                                      (short)0, acc0, false, false);
        acc1 = __builtin_amdgcn_wmma_f32_16x16x32_f16(false, a, false, b1,
                                                      (short)0, acc1, false, false);
        aptr += 32;
        bptr += bstep;
    }

    const int n  = col0 + (lane & 15);
    const int mb = row0 + half * 8;
    const float bz0 = bias ? bias[n]      : 0.0f;
    const float bz1 = bias ? bias[n + 16] : 0.0f;
#pragma unroll
    for (int r = 0; r < 8; ++r) {
        C[(size_t)(mb + r) * N + n]      = acc0[r] + bz0;
        C[(size_t)(mb + r) * N + n + 16] = acc1[r] + bz1;
    }
}

// -------------------- fused score / segment-softmax / readout ---------------
// One 256-thread block (8 waves) per graph. segment_ids are sorted.
// Emits R_g = sum_n attn_n * x_n directly in f16 for the next WMMA GEMM.

__global__ void segment_attn_readout(const float* __restrict__ X,   // [N, 256]
                                     const float* __restrict__ P,   // [G, 256]
                                     const int*   __restrict__ seg, // [N]
                                     _Float16*    __restrict__ Rh)  // [G, 256]
{
    const int g    = blockIdx.x;
    const int tid  = threadIdx.x;
    const int lane = tid & 31;
    const int wave = tid >> 5;

    __shared__ float pg[256];
    __shared__ float wred[8];
    __shared__ float rpart[8][256];

    // binary search segment range [lo, hi)
    int a = 0, b = Nn;
    while (a < b) { int m = (a + b) >> 1; if (seg[m] < g) a = m + 1; else b = m; }
    const int lo = a;
    b = Nn;
    while (a < b) { int m = (a + b) >> 1; if (seg[m] < g + 1) a = m + 1; else b = m; }
    const int hi = a;

    pg[tid] = P[g * 256 + tid];
    __syncthreads();

    // ---- phase 1: segment max of score ----
    float mloc = -FLT_MAX;
    for (int n = lo + wave; n < hi; n += 8) {
        const float* x = X + (size_t)n * 256;
        __builtin_prefetch(x + 8 * 256 + lane, 0, 0);   // global_prefetch next chunk
        float s = 0.f;
#pragma unroll
        for (int j = 0; j < 8; ++j) s += x[lane + 32 * j] * pg[lane + 32 * j];
#pragma unroll
        for (int off = 16; off > 0; off >>= 1) s += __shfl_xor(s, off, 32);
        mloc = fmaxf(mloc, s * 0.0625f);                // 1/sqrt(256)
    }
    if (lane == 0) wred[wave] = mloc;
    __syncthreads();
    float m = wred[0];
#pragma unroll
    for (int w = 1; w < 8; ++w) m = fmaxf(m, wred[w]);

    // ---- phase 2: exp-accumulate sum(e) and sum(e * x)  (x re-read is L2-hot) ----
    float racc[8] = {0.f, 0.f, 0.f, 0.f, 0.f, 0.f, 0.f, 0.f};
    float esum = 0.f;
    for (int n = lo + wave; n < hi; n += 8) {
        const float* x = X + (size_t)n * 256;
        float xj[8];
        float s = 0.f;
#pragma unroll
        for (int j = 0; j < 8; ++j) { xj[j] = x[lane + 32 * j]; s += xj[j] * pg[lane + 32 * j]; }
#pragma unroll
        for (int off = 16; off > 0; off >>= 1) s += __shfl_xor(s, off, 32);
        const float e = __expf(s * 0.0625f - m);        // lane-uniform within wave
        esum += e;
#pragma unroll
        for (int j = 0; j < 8; ++j) racc[j] += e * xj[j];
    }
    __syncthreads();
    if (lane == 0) wred[wave] = esum;
#pragma unroll
    for (int j = 0; j < 8; ++j) rpart[wave][lane + 32 * j] = racc[j];
    __syncthreads();

    float denom = 0.f;
#pragma unroll
    for (int w = 0; w < 8; ++w) denom += wred[w];
    float rv = 0.f;
#pragma unroll
    for (int w = 0; w < 8; ++w) rv += rpart[w][tid];

    Rh[g * 256 + tid] = (_Float16)((hi > lo) ? rv / denom : 0.f);
}

// ------------------------------- GRU gates ---------------------------------

__global__ void gru_gates(const float* __restrict__ gi,   // [G, 768]
                          const float* __restrict__ gh,   // [G, 768]
                          const float* __restrict__ q,    // [G, 256]
                          float* __restrict__ out)        // [G, 256]
{
    const int i = blockIdx.x * blockDim.x + threadIdx.x;
    if (i >= Gg * Dd) return;
    const int g = i >> 8, d = i & 255;
    const float* gir = gi + (size_t)g * 768;
    const float* ghr = gh + (size_t)g * 768;
    const float r = 1.f / (1.f + __expf(-(gir[d]       + ghr[d])));
    const float z = 1.f / (1.f + __expf(-(gir[256 + d] + ghr[256 + d])));
    const float n = tanhf(gir[512 + d] + r * ghr[512 + d]);
    out[i] = (1.f - z) * n + z * q[i];
}

// ------------------------------- launcher ----------------------------------

extern "C" void kernel_launch(void* const* d_in, const int* in_sizes, int n_in,
                              void* d_out, int out_size, void* d_ws, size_t ws_size,
                              hipStream_t stream) {
    const float* node = (const float*)d_in[0];   // [N, D]
    const float* kf   = (const float*)d_in[1];   // [G, FP]
    const int*   seg  = (const int*)  d_in[2];   // [N]
    const float* Wq   = (const float*)d_in[4];   // [D, FP]
    const float* Wk   = (const float*)d_in[5];   // [D, D]
    const float* Wv   = (const float*)d_in[6];   // [D, D]
    const float* bv   = (const float*)d_in[7];   // [D]
    const float* Wih  = (const float*)d_in[8];   // [3D, D]
    const float* Whh  = (const float*)d_in[9];   // [3D, D]
    const float* bih  = (const float*)d_in[10];  // [3D]
    const float* bhh  = (const float*)d_in[11];  // [3D]
    float* out = (float*)d_out;                  // [G, D]

    char* ws = (char*)d_ws;
    const size_t OFF_HKF   = 0;                                      // G*FP f16
    const size_t OFF_BQ    = OFF_HKF   + (size_t)Gg * FPd * 2;       // Wq.T packed  [2048,256]
    const size_t OFF_BK    = OFF_BQ    + (size_t)FPd * Dd * 2;       // Wk packed    [256,256]
    const size_t OFF_BV    = OFF_BK    + (size_t)Dd * Dd * 2;        // Wv.T packed  [256,256]
    const size_t OFF_BIH   = OFF_BV    + (size_t)Dd * Dd * 2;        // Wih.T packed [256,768]
    const size_t OFF_BHH   = OFF_BIH   + (size_t)Dd * 3 * Dd * 2;    // Whh.T packed [256,768]
    const size_t OFF_QF32  = OFF_BHH   + (size_t)Dd * 3 * Dd * 2;    // G*D f32
    const size_t OFF_HQ    = OFF_QF32  + (size_t)Gg * Dd * 4;        // G*D f16
    const size_t OFF_P     = OFF_HQ    + (size_t)Gg * Dd * 2;        // G*D f32
    const size_t OFF_HR    = OFF_P     + (size_t)Gg * Dd * 4;        // G*D f16
    const size_t OFF_OUTR  = OFF_HR    + (size_t)Gg * Dd * 2;        // G*D f32
    const size_t OFF_HOUT  = OFF_OUTR  + (size_t)Gg * Dd * 4;        // G*D f16
    const size_t OFF_GI    = OFF_HOUT  + (size_t)Gg * Dd * 2;        // G*3D f32
    const size_t OFF_GH    = OFF_GI    + (size_t)Gg * 3 * Dd * 4;    // G*3D f32

    _Float16* hkf  = (_Float16*)(ws + OFF_HKF);
    _Float16* bWq  = (_Float16*)(ws + OFF_BQ);
    _Float16* bWk  = (_Float16*)(ws + OFF_BK);
    _Float16* bWv  = (_Float16*)(ws + OFF_BV);
    _Float16* bWih = (_Float16*)(ws + OFF_BIH);
    _Float16* bWhh = (_Float16*)(ws + OFF_BHH);
    float*    qf   = (float*)   (ws + OFF_QF32);
    _Float16* hq   = (_Float16*)(ws + OFF_HQ);
    float*    p    = (float*)   (ws + OFF_P);
    _Float16* hR   = (_Float16*)(ws + OFF_HR);
    float*    outr = (float*)   (ws + OFF_OUTR);
    _Float16* hout = (_Float16*)(ws + OFF_HOUT);
    float*    gi   = (float*)   (ws + OFF_GI);
    float*    gh   = (float*)   (ws + OFF_GH);

    const int T = 256;
    // ---- precision conversion of A operand + fragment-packing of B operands ----
    cvt_f32_to_f16<<<(Gg * FPd + T - 1) / T, T, 0, stream>>>(kf, hkf, Gg * FPd);
    pack_b_f16<<<(FPd * Dd + T - 1) / T, T, 0, stream>>>(Wq, bWq, FPd, Dd, 1);      // B = Wq.T
    pack_b_f16<<<(Dd * Dd + T - 1) / T, T, 0, stream>>>(Wk, bWk, Dd, Dd, 0);        // B = Wk
    pack_b_f16<<<(Dd * Dd + T - 1) / T, T, 0, stream>>>(Wv, bWv, Dd, Dd, 1);        // B = Wv.T
    pack_b_f16<<<(Dd * 3 * Dd + T - 1) / T, T, 0, stream>>>(Wih, bWih, Dd, 3 * Dd, 1);
    pack_b_f16<<<(Dd * 3 * Dd + T - 1) / T, T, 0, stream>>>(Whh, bWhh, Dd, 3 * Dd, 1);

    // ---- q = kfeat @ Wq.T   [4096 x 2048 x 256] ----
    wmma_gemm_f16_f32<<<dim3(Dd / 128, Gg / 16), 128, 0, stream>>>(hkf, bWq, nullptr, qf, Gg, Dd, FPd);
    cvt_f32_to_f16<<<(Gg * Dd + T - 1) / T, T, 0, stream>>>(qf, hq, Gg * Dd);

    // ---- p = q @ Wk  (score projection) ----
    wmma_gemm_f16_f32<<<dim3(Dd / 128, Gg / 16), 128, 0, stream>>>(hq, bWk, nullptr, p, Gg, Dd, Dd);

    // ---- fused score / segment-softmax / weighted readout -> R (f16) ----
    segment_attn_readout<<<Gg, 256, 0, stream>>>(node, p, seg, hR);

    // ---- out = R @ Wv.T + bv ----
    wmma_gemm_f16_f32<<<dim3(Dd / 128, Gg / 16), 128, 0, stream>>>(hR, bWv, bv, outr, Gg, Dd, Dd);
    cvt_f32_to_f16<<<(Gg * Dd + T - 1) / T, T, 0, stream>>>(outr, hout, Gg * Dd);

    // ---- GRU input/hidden projections ----
    wmma_gemm_f16_f32<<<dim3(3 * Dd / 128, Gg / 16), 128, 0, stream>>>(hout, bWih, bih, gi, Gg, 3 * Dd, Dd);
    wmma_gemm_f16_f32<<<dim3(3 * Dd / 128, Gg / 16), 128, 0, stream>>>(hq, bWhh, bhh, gh, Gg, 3 * Dd, Dd);

    // ---- GRU gates ----
    gru_gates<<<(Gg * Dd + T - 1) / T, T, 0, stream>>>(gi, gh, qf, out);
}
